// PDAM_15298673508639
// MI455X (gfx1250) — matrix operand order repaired and checked by reading, not compiled, not run on hardware
//
#include <hip/hip_runtime.h>
#include <math.h>

#define B_ 16
#define S_ 1024
#define D_ 128
#define NTOT (B_ * S_ * D_)
#define MROWS (B_ * S_)
#define NOISE_STD_ 0.1f
#define TOL_ 1e-3f
#define QSCALE_ 0.0883883476483184405f  // 1/sqrt(128)

typedef __attribute__((ext_vector_type(16))) _Float16 v16h;
typedef __attribute__((ext_vector_type(8)))  _Float16 v8h;
typedef __attribute__((ext_vector_type(8)))  float    v8f;
typedef __attribute__((ext_vector_type(4)))  unsigned v4u;
typedef __attribute__((ext_vector_type(4)))  int      v4i;
typedef __attribute__((ext_vector_type(8)))  int      v8i;

// ---------------------------------------------------------------------------
// CDNA5 primitives
// ---------------------------------------------------------------------------
static __device__ __forceinline__ v8f wmma16(v16h a, v16h b, v8f c) {
  return __builtin_amdgcn_wmma_f32_16x16x32_f16(
      /*neg_a=*/false, a, /*neg_b=*/false, b,
      /*c_mod=*/(short)0, c, /*reuse_a=*/false, /*reuse_b=*/false);
}

// Per-lane async DMA: global -> LDS, 16B, tracked by ASYNCcnt.
static __device__ __forceinline__ void async_ld_b128(unsigned lds_byte_off,
                                                     const void* gbase,
                                                     int gbyte_off) {
  asm volatile("global_load_async_to_lds_b128 %0, %1, %2"
               :
               : "v"(lds_byte_off), "v"(gbyte_off), "s"(gbase)
               : "memory");
}
static __device__ __forceinline__ void wait_async0() {
  asm volatile("s_wait_asynccnt 0" ::: "memory");
}
static __device__ __forceinline__ void wait_async2() {
  asm volatile("s_wait_asynccnt 2" ::: "memory");
}

// LDS byte offset of a __shared__ pointer (low 32 bits of generic address).
template <typename T>
static __device__ __forceinline__ unsigned lds_off(T* p) {
  return (unsigned)(unsigned long long)(void*)p;
}

// Tensor Data Mover: load a 2D f16 tile (tile_d0 x tile_d1 elements, row
// stride stride0 elements) packed contiguously into LDS at lds_byte.
// D# group0/group1 packed per CDNA5 ISA 8.3/8.4.  tensor dims == tile dims
// (global_addr is the tile start, so OOB clipping must never trigger).
// flags: extra bits for group1 word0 (pad_enable/pad_interval/pad_amount).
static __device__ __forceinline__ void tdm_load_2d_f16(
    unsigned lds_byte, const void* gtile, unsigned tile_d0, unsigned tile_d1,
    unsigned tensor_d0, unsigned tensor_d1, unsigned stride0_elems,
    unsigned flags) {
  unsigned long long ga = (unsigned long long)gtile;
  v4u g0;
  g0[0] = 1u;                                     // count=1 valid descriptor
  g0[1] = lds_byte;                               // lds_addr
  g0[2] = (unsigned)(ga & 0xFFFFFFFFu);           // global_addr[31:0]
  g0[3] = (unsigned)((ga >> 32) & 0x01FFFFFFu) | (2u << 30);  // [56:32]|type=2
  v8i g1;
  g1[0] = (int)((1u << 16) | flags);              // data_size=1 (2B) + flags
  g1[1] = (int)((tensor_d0 & 0xFFFFu) << 16);     // tensor_dim0[15:0] @ b63:48
  g1[2] = (int)(((tensor_d0 >> 16) & 0xFFFFu) |   // tensor_dim0[31:16]
                ((tensor_d1 & 0xFFFFu) << 16));   // tensor_dim1[15:0]
  g1[3] = (int)(((tensor_d1 >> 16) & 0xFFFFu) |   // tensor_dim1[31:16]
                ((tile_d0 & 0xFFFFu) << 16));     // tile_dim0
  g1[4] = (int)(tile_d1 & 0xFFFFu);               // tile_dim1 (tile_dim2=0)
  g1[5] = (int)(stride0_elems);                   // tensor_dim0_stride[31:0]
  g1[6] = 0;                                      // stride0[47:32]|stride1 lo
  g1[7] = 0;                                      // stride1 hi (2D: unused)
  v4i gz4 = {0, 0, 0, 0};
  v8i gz8 = {0, 0, 0, 0, 0, 0, 0, 0};
  __builtin_amdgcn_tensor_load_to_lds(g0, g1, gz4, gz4, gz8, 0);
}

// A-fragment 16x32 f16, row-major tile, 16B-aligned rows -> 2x ds_load_b128.
static __device__ __forceinline__ v16h frag_A(const _Float16* tile, int ld) {
  const int lane = threadIdx.x & 31;
  const int m = lane & 15, half = lane >> 4;
  const _Float16* r0 = tile + m * ld + half * 8;
  union { v16h v; v8h h[2]; } u;
  u.h[0] = *(const v8h*)(r0);
  u.h[1] = *(const v8h*)(r0 + 16);
  return u.v;
}

// B-fragment 32x16 f16 stored N-major -> 2x ds_load_b128.
static __device__ __forceinline__ v16h frag_Bn(const _Float16* tile, int ld) {
  const int lane = threadIdx.x & 31;
  const int n = lane & 15, half = lane >> 4;
  const _Float16* r0 = tile + n * ld + half * 8;
  union { v16h v; v8h h[2]; } u;
  u.h[0] = *(const v8h*)(r0);
  u.h[1] = *(const v8h*)(r0 + 16);
  return u.v;
}

// Deterministic Gaussian noise (splitmix64 + Box-Muller).
static __device__ __forceinline__ float gnoise(unsigned seed, unsigned idx) {
  unsigned long long z = (((unsigned long long)seed) << 32) ^ (unsigned long long)idx;
  z += 0x9E3779B97F4A7C15ULL;
  z = (z ^ (z >> 30)) * 0xBF58476D1CE4E5B9ULL;
  z = (z ^ (z >> 27)) * 0x94D049BB133111EBULL;
  z ^= (z >> 31);
  unsigned u0 = (unsigned)z, u1 = (unsigned)(z >> 32);
  float r  = sqrtf(-2.0f * __logf((u0 + 1.0f) * 2.3283064365386963e-10f));
  float th = 6.28318530717958647f * (u1 * 2.3283064365386963e-10f);
  return r * __cosf(th);
}

// ---------------------------------------------------------------------------
// Prep kernels
// ---------------------------------------------------------------------------
__global__ void cast_f16_k(const float* __restrict__ x, _Float16* __restrict__ y,
                           int n) {
  int i = blockIdx.x * blockDim.x + threadIdx.x;
  if (i < n) y[i] = (_Float16)x[i];
}

__global__ void transpose_w16_k(const float* __restrict__ W,
                                _Float16* __restrict__ Wt) {
  int idx = blockIdx.x * blockDim.x + threadIdx.x;  // 128*128
  if (idx >= 128 * 128) return;
  int n = idx >> 7, k = idx & 127;
  Wt[idx] = (_Float16)W[k * 128 + n];
}

// ---------------------------------------------------------------------------
// Projection: Yh = (Xh @ W + b + sigma*noise) * outscale, f16 out, optional
// transposed output for V.  W staged once by TDM (with DMA-side LDS padding);
// X tiles double-buffered via async DMA (s_wait_asynccnt 2 pipelining).
// ---------------------------------------------------------------------------
__global__ __launch_bounds__(256) void proj_gemm_wmma(
    const _Float16* __restrict__ Xh, const _Float16* __restrict__ Wt,
    const float* __restrict__ bias, _Float16* __restrict__ Yh,
    unsigned seed, float sigma, float outscale, int transposeOut) {
  __shared__ alignas(16) _Float16 Xs[2][128 * 40];  // ping-pong X tiles, ld 40
  __shared__ alignas(16) _Float16 Wts[128 * 136];   // 128 n x 128 k, ld 136
  const int tid = threadIdx.x, wave = tid >> 5, lane = tid & 31;
  const int half = lane >> 4, nloc = lane & 15;
  const int m0 = blockIdx.x * 128;
  const unsigned xs_b[2] = {lds_off(&Xs[0][0]), lds_off(&Xs[1][0])};

  // Wave 0: whole Wt via TDM; pad 4 DWORDs after every 64 DWORDs -> ld 136.
  if (tid < 32) {
    tdm_load_2d_f16(lds_off(Wts), Wt, 128, 128, 128, 128, 128,
                    (1u << 20) /*pad_en*/ | (5u << 22) /*64DW*/ |
                    (3u << 25) /*+4DW*/);
  }

  // each wave issues exactly 2 async ops per staged tile
  auto stageX = [&](int buf, int kk) {
    for (int c = tid; c < 128 * 4; c += 256) {
      int r = c >> 2, j = c & 3;
      async_ld_b128(xs_b[buf] + (unsigned)(r * 80 + j * 16), Xh,
                    (m0 + r) * 256 + kk * 2 + j * 16);
    }
  };
  stageX(0, 0);

  v8f zero = {};
  v8f acc[8];
#pragma unroll
  for (int i = 0; i < 8; ++i) acc[i] = zero;

  for (int st = 0; st < 4; ++st) {
    const int kk = st * 32;
    if (st < 3) {
      stageX((st + 1) & 1, kk + 32);  // prefetch next tile
      wait_async2();                  // <=2 outstanding => current tile landed
    } else {
      wait_async0();
    }
    if (st == 0 && tid < 32) __builtin_amdgcn_s_wait_tensorcnt(0);
    __syncthreads();

    const _Float16* Xcur = &Xs[st & 1][0];
    v16h a = frag_A(&Xcur[(wave * 16) * 40], 40);
    v16h bfr[8];
#pragma unroll
    for (int nt = 0; nt < 8; ++nt)
      bfr[nt] = frag_Bn(&Wts[(nt * 16) * 136 + kk], 136);
#pragma unroll
    for (int nt = 0; nt < 8; ++nt) acc[nt] = wmma16(a, bfr[nt], acc[nt]);
    __syncthreads();  // done reading Xs[st&1] before it is restaged at st+2
  }
#pragma unroll
  for (int nt = 0; nt < 8; ++nt) {
#pragma unroll
    for (int r = 0; r < 8; ++r) {
      int m = m0 + wave * 16 + r + 8 * half;
      int n = nt * 16 + nloc;
      float o = acc[nt][r] + bias[n] + sigma * gnoise(seed, (unsigned)(m * 128 + n));
      o *= outscale;
      if (transposeOut) {
        int bb = m >> 10, s = m & (S_ - 1);
        Yh[((long)bb * D_ + n) * S_ + s] = (_Float16)o;
      } else {
        Yh[(long)m * D_ + n] = (_Float16)o;
      }
    }
  }
}

// ---------------------------------------------------------------------------
// Flash attention.  Qh pre-scaled f16 [B,S,D]; Kh f16 [B,S,D]; Vth f16 [B,D,S].
// Q staged by async DMA, fragments pinned in VGPRs; K/V tiles double-buffered
// via TDM with s_wait_tensorcnt(2) pipelining (DMA overlaps WMMA compute).
// Per 32KB buffer: K tile at +0 (64x128), V^T tile at +8192 f16 (128x64).
// ---------------------------------------------------------------------------
__global__ __launch_bounds__(128) void flash_attn_wmma(
    const _Float16* __restrict__ Qh, const _Float16* __restrict__ Kh,
    const _Float16* __restrict__ Vth, float* __restrict__ Og) {
  __shared__ alignas(16) _Float16 KV[2][16384];    // 2 x (Ks 8192 | Vts 8192)
  __shared__ alignas(16) _Float16 Ps[4][16 * 64];  // wave-private P, [m][key]

  const int tid = threadIdx.x, wave = tid >> 5, lane = tid & 31;
  const int half = lane >> 4, nloc = lane & 15;
  const int b = blockIdx.y, q0 = blockIdx.x * 64;
  const unsigned kv_b[2] = {lds_off(&KV[0][0]), lds_off(&KV[1][0])};

  // stage Q tile into KV[1] via async DMA, pin fragments, then recycle KV[1]
  {
    const _Float16* Qp = Qh + ((long)b * S_ + q0) * D_;
    for (int c = tid; c < 1024; c += 128)
      async_ld_b128(kv_b[1] + c * 16, Qp, c * 16);
  }
  wait_async0();
  __syncthreads();
  v16h qfr[4];
#pragma unroll
  for (int kk = 0; kk < 4; ++kk)
    qfr[kk] = frag_A(&KV[1][(wave * 16) * 128 + kk * 32], 128);

  v8f zero = {};
  v8f oacc[8];
#pragma unroll
  for (int i = 0; i < 8; ++i) oacc[i] = zero;
  float mrow[8], lrow[8];
#pragma unroll
  for (int r = 0; r < 8; ++r) { mrow[r] = -1e30f; lrow[r] = 0.f; }
  __syncthreads();  // everyone pinned Q before KV[1] is overwritten

  auto issue_tile = [&](int kt, int buf) {
    tdm_load_2d_f16(kv_b[buf], Kh + ((long)b * S_ + kt * 64) * D_,
                    128, 64, 128, 64, 128, 0);
    tdm_load_2d_f16(kv_b[buf] + 16384, Vth + (long)b * D_ * S_ + kt * 64,
                    64, 128, 64, 128, S_, 0);
  };
  if (tid < 32) issue_tile(0, 0);

  const int NT = S_ / 64;
  for (int kt = 0; kt < NT; ++kt) {
    const int cur = kt & 1;
    if (tid < 32) {
      if (kt + 1 < NT) {
        issue_tile(kt + 1, cur ^ 1);            // prefetch next K/V pair
        __builtin_amdgcn_s_wait_tensorcnt(2);   // <=2 outstanding => tile kt in
      } else {
        __builtin_amdgcn_s_wait_tensorcnt(0);
      }
    }
    __syncthreads();
    const _Float16* Ks  = &KV[cur][0];
    const _Float16* Vts = &KV[cur][8192];

    // S = Q K^T : 16 q-rows x 64 keys per wave (4 n-tiles x 4 k-steps)
    v8f sacc[4];
#pragma unroll
    for (int nt = 0; nt < 4; ++nt) sacc[nt] = zero;
#pragma unroll
    for (int kk = 0; kk < 4; ++kk) {
      v16h bfr[4];
#pragma unroll
      for (int nt = 0; nt < 4; ++nt)
        bfr[nt] = frag_Bn(&Ks[(nt * 16) * 128 + kk * 32], 128);
#pragma unroll
      for (int nt = 0; nt < 4; ++nt)
        sacc[nt] = wmma16(qfr[kk], bfr[nt], sacc[nt]);
    }

    // online softmax (row M = r + 8*half lives in 16 lanes of one half)
    float mnew[8], rsum[8];
#pragma unroll
    for (int r = 0; r < 8; ++r) {
      float v = fmaxf(fmaxf(sacc[0][r], sacc[1][r]),
                      fmaxf(sacc[2][r], sacc[3][r]));
#pragma unroll
      for (int off = 8; off > 0; off >>= 1) v = fmaxf(v, __shfl_xor(v, off, 32));
      mnew[r] = fmaxf(mrow[r], v);
      rsum[r] = 0.f;
    }
#pragma unroll
    for (int nt = 0; nt < 4; ++nt)
#pragma unroll
      for (int r = 0; r < 8; ++r) {
        float p = __expf(sacc[nt][r] - mnew[r]);
        sacc[nt][r] = p;
        rsum[r] += p;
      }
#pragma unroll
    for (int r = 0; r < 8; ++r) {
      float s = rsum[r];
#pragma unroll
      for (int off = 8; off > 0; off >>= 1) s += __shfl_xor(s, off, 32);
      float alpha = __expf(mrow[r] - mnew[r]);
      lrow[r] = lrow[r] * alpha + s;
      mrow[r] = mnew[r];
#pragma unroll
      for (int nt = 0; nt < 8; ++nt) oacc[nt][r] *= alpha;
    }

    // spill P (C layout) into wave-private LDS, reload as A-fragments
#pragma unroll
    for (int nt = 0; nt < 4; ++nt)
#pragma unroll
      for (int r = 0; r < 8; ++r)
        Ps[wave][(r + 8 * half) * 64 + nt * 16 + nloc] = (_Float16)sacc[nt][r];

#pragma unroll
    for (int kk = 0; kk < 2; ++kk) {
      v16h a = frag_A(&Ps[wave][kk * 32], 64);
      v16h bfr[8];
#pragma unroll
      for (int nt = 0; nt < 8; ++nt)
        bfr[nt] = frag_Bn(&Vts[(nt * 16) * 64 + kk * 32], 64);
#pragma unroll
      for (int nt = 0; nt < 8; ++nt) oacc[nt] = wmma16(a, bfr[nt], oacc[nt]);
    }
    __syncthreads();  // reads of KV[cur] done before it is restaged at kt+2
  }

#pragma unroll
  for (int nt = 0; nt < 8; ++nt)
#pragma unroll
    for (int r = 0; r < 8; ++r) {
      int m = wave * 16 + r + 8 * half;
      int n = nt * 16 + nloc;
      Og[((long)b * S_ + q0 + m) * D_ + n] = oacc[nt][r] / lrow[r];
    }
}

// ---------------------------------------------------------------------------
// UNet building blocks (f32 direct; ~22% of total FLOPs)
// ---------------------------------------------------------------------------
__global__ void conv1d_k(const float* __restrict__ x, const float* __restrict__ w,
                         const float* __restrict__ bias, float* __restrict__ y,
                         int Bn, int L, int Ci, int Co, int KW, int relu) {
  int idx = blockIdx.x * blockDim.x + threadIdx.x;
  int total = Bn * L * Co;
  if (idx >= total) return;
  int co = idx % Co;
  int s  = (idx / Co) % L;
  int b  = idx / (Co * L);
  float acc = bias[co];
  int pad = (KW - 1) / 2;
  for (int t = 0; t < KW; ++t) {
    int sp = s + t - pad;
    if (sp < 0 || sp >= L) continue;
    const float* xp = x + ((long)b * L + sp) * Ci;
    const float* wp = w + (long)t * Ci * Co + co;
    __builtin_prefetch(xp + Ci, 0, 0);  // next tap's input row
    for (int ci = 0; ci < Ci; ++ci) acc = fmaf(xp[ci], wp[(long)ci * Co], acc);
  }
  if (relu) acc = fmaxf(acc, 0.f);
  y[idx] = acc;
}

__global__ void convT1d_k(const float* __restrict__ x, const float* __restrict__ w,
                          const float* __restrict__ bias, float* __restrict__ y,
                          int Bn, int Lin, int Ci, int Co) {
  int Lout = 2 * Lin;
  int idx = blockIdx.x * blockDim.x + threadIdx.x;
  int total = Bn * Lout * Co;
  if (idx >= total) return;
  int co = idx % Co;
  int o  = (idx / Co) % Lout;
  int b  = idx / (Co * Lout);
  float acc = bias[co];
  for (int t = 0; t < 3; ++t) {
    int j = o + t - 1;
    if (j < 0 || (j & 1)) continue;
    int i = j >> 1;
    if (i >= Lin) continue;
    const float* xp = x + ((long)b * Lin + i) * Ci;
    const float* wp = w + (long)t * Ci * Co + co;
    for (int ci = 0; ci < Ci; ++ci) acc = fmaf(xp[ci], wp[(long)ci * Co], acc);
  }
  y[idx] = fmaxf(acc, 0.f);
}

__global__ void pool2_k(const float* __restrict__ x, float* __restrict__ y,
                        int Bn, int Lout, int C) {
  int idx = blockIdx.x * blockDim.x + threadIdx.x;
  int total = Bn * Lout * C;
  if (idx >= total) return;
  int c = idx % C;
  int s = (idx / C) % Lout;
  int b = idx / (C * Lout);
  const float* xp = x + ((long)b * (2 * Lout) + 2 * s) * C + c;
  y[idx] = fmaxf(xp[0], xp[C]);
}

__global__ void repeat2_k(const float* __restrict__ x, float* __restrict__ y,
                          int Bn, int Lin, int C) {
  int Lout = 2 * Lin;
  int idx = blockIdx.x * blockDim.x + threadIdx.x;
  int total = Bn * Lout * C;
  if (idx >= total) return;
  int c = idx % C;
  int s = (idx / C) % Lout;
  int b = idx / (C * Lout);
  y[idx] = x[((long)b * Lin + (s >> 1)) * C + c];
}

__global__ void concat_crop_k(const float* __restrict__ a, int La,
                              const float* __restrict__ e, float* __restrict__ y,
                              int Bn, int L, int Ca, int Ce, int cl) {
  int C = Ca + Ce;
  int idx = blockIdx.x * blockDim.x + threadIdx.x;
  int total = Bn * L * C;
  if (idx >= total) return;
  int c = idx % C;
  int s = (idx / C) % L;
  int b = idx / (C * L);
  float v = (c < Ca) ? a[((long)b * La + s + cl) * Ca + c]
                     : e[((long)b * L + s) * Ce + (c - Ca)];
  y[idx] = v;
}

__global__ void softmax_lastdim_k(const float* __restrict__ x,
                                  float* __restrict__ y, int rows, int C) {
  int r = blockIdx.x * blockDim.x + threadIdx.x;
  if (r >= rows) return;
  const float* xp = x + (long)r * C;
  float* yp = y + (long)r * C;
  float mx = -1e30f;
  for (int c = 0; c < C; ++c) mx = fmaxf(mx, xp[c]);
  float s = 0.f;
  for (int c = 0; c < C; ++c) { float e = __expf(xp[c] - mx); yp[c] = e; s += e; }
  float inv = 1.0f / s;
  for (int c = 0; c < C; ++c) yp[c] *= inv;
}

// ---------------------------------------------------------------------------
// while-loop machinery
// ---------------------------------------------------------------------------
__global__ void fill_zero_k(float* p, int n) {
  int i = blockIdx.x * blockDim.x + threadIdx.x;
  if (i < n) p[i] = 0.f;
}

__global__ void init_scalars_k(float* acc, int* active, int setActive) {
  *acc = 0.f;
  if (setActive) *active = 1;
}

__global__ void reduce_absdiff_k(const float* __restrict__ a,
                                 const float* __restrict__ b,
                                 float* acc, int n) {
  float s = 0.f;
  for (int i = blockIdx.x * blockDim.x + threadIdx.x; i < n;
       i += gridDim.x * blockDim.x)
    s += fabsf(a[i] - b[i]);
#pragma unroll
  for (int off = 16; off > 0; off >>= 1) s += __shfl_xor(s, off, 32);
  if ((threadIdx.x & 31) == 0) atomicAdd(acc, s);
}

__global__ void eval_cond_k(const float* acc, int* active, float invn) {
  if ((*active != 0) && ((*acc) * invn > TOL_)) *active = 1;
  else *active = 0;
}

__global__ void update_state_k(const int* __restrict__ condp,
                               const float* __restrict__ Zi,
                               float* __restrict__ Z, float* __restrict__ prev,
                               _Float16* __restrict__ Qh,
                               _Float16* __restrict__ Kh,
                               _Float16* __restrict__ Vth, int n, int it) {
  if (*condp == 0) return;
  int i = blockIdx.x * blockDim.x + threadIdx.x;
  if (i >= n) return;
  float zi = Zi[i];
  prev[i] = Z[i];
  Z[i] = zi;
  float qn = zi + NOISE_STD_ * gnoise((unsigned)(10 + 3 * it), (unsigned)i);
  float kn = zi + NOISE_STD_ * gnoise((unsigned)(11 + 3 * it), (unsigned)i);
  float vn = zi + NOISE_STD_ * gnoise((unsigned)(12 + 3 * it), (unsigned)i);
  Qh[i] = (_Float16)(qn * QSCALE_);
  Kh[i] = (_Float16)kn;
  int d = i & (D_ - 1), s = (i >> 7) & (S_ - 1), b = i >> 17;
  Vth[((long)b * D_ + d) * S_ + s] = (_Float16)vn;
}

// ---------------------------------------------------------------------------
// Host orchestration
// ---------------------------------------------------------------------------
static inline int cdiv(int a, int b) { return (a + b - 1) / b; }

static void launch_unet(const float* Zi, float* out, void* const* d_in,
                        float* t0, float* t1, float* t2, float* t3, float* t4,
                        float* t5, hipStream_t stream) {
  const float** P = (const float**)d_in;
  const int TB = 256;
#define CONV(x, wi, y, Bn, L, Ci, Co, KW, relu) \
  conv1d_k<<<cdiv((Bn) * (L) * (Co), TB), TB, 0, stream>>>( \
      x, P[wi], P[(wi) + 1], y, Bn, L, Ci, Co, KW, relu)
#define CONVT(x, wi, y, Bn, Lin, Ci, Co) \
  convT1d_k<<<cdiv((Bn) * 2 * (Lin) * (Co), TB), TB, 0, stream>>>( \
      x, P[wi], P[(wi) + 1], y, Bn, Lin, Ci, Co)
  CONV(Zi, 8, t3, B_, S_, 128, 32, 3, 1);
  CONV(t3, 10, t0, B_, S_, 32, 32, 3, 1);                        // e1 = t0
  pool2_k<<<cdiv(B_ * (S_ / 2) * 32, TB), TB, 0, stream>>>(t0, t3, B_, S_ / 2, 32);
  CONV(t3, 12, t4, B_, S_ / 2, 32, 64, 3, 1);
  CONV(t4, 14, t1, B_, S_ / 2, 64, 64, 3, 1);                    // e2 = t1
  pool2_k<<<cdiv(B_ * (S_ / 4) * 64, TB), TB, 0, stream>>>(t1, t3, B_, S_ / 4, 64);
  CONV(t3, 16, t4, B_, S_ / 4, 64, 128, 3, 1);
  CONV(t4, 18, t2, B_, S_ / 4, 128, 128, 3, 1);                  // m = t2
  repeat2_k<<<cdiv(B_ * (S_ / 2) * 128, TB), TB, 0, stream>>>(t2, t3, B_, S_ / 4, 128);
  CONVT(t3, 20, t4, B_, S_ / 2, 128, 64);                        // [B,S,64]
  CONV(t4, 22, t5, B_, S_, 64, 64, 3, 1);                        // d2 [B,S,64]
  concat_crop_k<<<cdiv(B_ * (S_ / 2) * 128, TB), TB, 0, stream>>>(
      t5, S_, t1, t3, B_, S_ / 2, 64, 64, (S_ - S_ / 2) / 2);    // [B,S/2,128]
  repeat2_k<<<cdiv(B_ * S_ * 128, TB), TB, 0, stream>>>(t3, t4, B_, S_ / 2, 128);
  CONVT(t4, 24, t5, B_, S_, 128, 32);                            // [B,2S,32]
  CONV(t5, 26, t4, B_, 2 * S_, 32, 32, 3, 1);                    // d1 [B,2S,32]
  concat_crop_k<<<cdiv(B_ * S_ * 64, TB), TB, 0, stream>>>(
      t4, 2 * S_, t0, t5, B_, S_, 32, 32, (2 * S_ - S_) / 2);    // [B,S,64]
  CONV(t5, 28, t4, B_, S_, 64, 128, 1, 0);                       // [B,S,128]
  softmax_lastdim_k<<<cdiv(MROWS, TB), TB, 0, stream>>>(t4, out, MROWS, 128);
#undef CONV
#undef CONVT
}

extern "C" void kernel_launch(void* const* d_in, const int* in_sizes, int n_in,
                              void* d_out, int out_size, void* d_ws, size_t ws_size,
                              hipStream_t stream) {
  (void)in_sizes; (void)n_in; (void)out_size; (void)ws_size;
  const float* x1 = (const float*)d_in[0];
  const float* x2 = (const float*)d_in[1];
  const float* Wq = (const float*)d_in[2]; const float* bq = (const float*)d_in[3];
  const float* Wk = (const float*)d_in[4]; const float* bk = (const float*)d_in[5];
  const float* Wv = (const float*)d_in[6]; const float* bv = (const float*)d_in[7];

  float* ws = (float*)d_ws;
  float* Z    = ws + 0L * NTOT;
  float* prev = ws + 1L * NTOT;
  float* Zi   = ws + 2L * NTOT;
  float* t0   = ws + 3L * NTOT;
  float* t1   = ws + 4L * NTOT;
  float* t2   = ws + 5L * NTOT;
  float* t3   = ws + 6L * NTOT;
  float* t4   = ws + 7L * NTOT;
  float* t5   = ws + 8L * NTOT;
  _Float16* h16 = (_Float16*)(ws + 9L * NTOT);
  _Float16* Qh  = h16 + 0L * NTOT;
  _Float16* Kh  = h16 + 1L * NTOT;
  _Float16* Vth = h16 + 2L * NTOT;
  _Float16* x1h = h16 + 3L * NTOT;
  _Float16* x2h = h16 + 4L * NTOT;
  _Float16* Wqt = h16 + 5L * NTOT;
  _Float16* Wkt = Wqt + 128 * 128;
  _Float16* Wvt = Wkt + 128 * 128;
  float* accp = (float*)(Wvt + 128 * 128 + 64);
  int*   actp = (int*)(accp + 1);

  const int TB = 256;
  const dim3 gGemm(MROWS / 128), bGemm(256);
  const dim3 gAttn(S_ / 64, B_), bAttn(128);

  cast_f16_k<<<cdiv(NTOT, TB), TB, 0, stream>>>(x1, x1h, NTOT);
  cast_f16_k<<<cdiv(NTOT, TB), TB, 0, stream>>>(x2, x2h, NTOT);
  transpose_w16_k<<<64, 256, 0, stream>>>(Wq, Wqt);
  transpose_w16_k<<<64, 256, 0, stream>>>(Wk, Wkt);
  transpose_w16_k<<<64, 256, 0, stream>>>(Wv, Wvt);

  proj_gemm_wmma<<<gGemm, bGemm, 0, stream>>>(x1h, Wqt, bq, Qh, 0u, NOISE_STD_,
                                              QSCALE_, 0);
  proj_gemm_wmma<<<gGemm, bGemm, 0, stream>>>(x1h, Wkt, bk, Kh, 1u, NOISE_STD_,
                                              1.0f, 0);
  proj_gemm_wmma<<<gGemm, bGemm, 0, stream>>>(x2h, Wvt, bv, Vth, 2u, NOISE_STD_,
                                              1.0f, 1);

  fill_zero_k<<<cdiv(NTOT, TB), TB, 0, stream>>>(prev, NTOT);
  init_scalars_k<<<1, 1, 0, stream>>>(accp, actp, 1);

  flash_attn_wmma<<<gAttn, bAttn, 0, stream>>>(Qh, Kh, Vth, Z);

  for (int it = 0; it < 5; ++it) {
    init_scalars_k<<<1, 1, 0, stream>>>(accp, actp, 0);
    reduce_absdiff_k<<<512, 256, 0, stream>>>(Z, prev, accp, NTOT);
    eval_cond_k<<<1, 1, 0, stream>>>(accp, actp, 1.0f / (float)NTOT);

    flash_attn_wmma<<<gAttn, bAttn, 0, stream>>>(Qh, Kh, Vth, Zi);
    launch_unet(Zi, Zi, d_in, t0, t1, t2, t3, t4, t5, stream);

    update_state_k<<<cdiv(NTOT, TB), TB, 0, stream>>>(actp, Zi, Z, prev, Qh, Kh,
                                                      Vth, NTOT, it);
  }

  (void)hipMemcpyAsync(d_out, Z, (size_t)NTOT * sizeof(float),
                       hipMemcpyDeviceToDevice, stream);
}
